// GRU_25726854103481
// MI455X (gfx1250) — matrix operand order repaired and checked by reading, not compiled, not run on hardware
//
#include <hip/hip_runtime.h>
#include <hip/hip_bf16.h>

// ---------------------------------------------------------------------------
// GRU for MI455X (gfx1250): bf16 WMMA GEMMs + fused gate epilogues.
//   Phase 1: pack weights to WMMA-fragment order, convert x to bf16,
//            one big GEMM for all input projections (xr/xz/xn), f32 out.
//   Phase 2: 512 per-timestep kernels; each wave owns a 16x16 output tile,
//            runs 3 gate GEMMs (K=1024) reusing the A fragment, then fuses
//            sigmoid/sigmoid/tanh and the h update. h is double-buffered.
//   The k-loops are software-pipelined: fragments for iteration kc+1 are
//   issued before the WMMAs of iteration kc, so the compiler's s_wait_loadcnt
//   becomes partial instead of a full drain before every WMMA.
// ---------------------------------------------------------------------------

typedef __bf16 bf16_t;
typedef __attribute__((ext_vector_type(16))) __bf16 v16bf;
typedef __attribute__((ext_vector_type(8)))  __bf16 v8bf;
typedef __attribute__((ext_vector_type(8)))  float  v8f;

#define B_  64
#define T_  512
#define D_  1024
#define H_  1024

// ---------------------------------------------------------------------------
// Pack a KxN f32 row-major weight matrix into bf16 WMMA B-fragment order.
// Fragment f = nTile*(K/32) + kChunk holds a 32x16 tile; within a fragment,
// lane L owns column N = nTile*16 + (L&15) and K-run (L>>4)*16 + j (j=0..15),
// stored contiguously so the GEMM does one 32B load per lane per fragment.
// ---------------------------------------------------------------------------
__global__ void pack_w_kernel(const float* __restrict__ W,
                              bf16_t* __restrict__ P, int K, int N) {
    int idx = blockIdx.x * blockDim.x + threadIdx.x;
    if (idx >= K * N) return;
    int KC    = K >> 5;            // k-chunks of 32
    int f     = idx >> 9;          // fragment index (512 elems each)
    int r     = idx & 511;
    int lane  = r >> 4;
    int j     = r & 15;
    int nTile = f / KC;
    int kc    = f - nTile * KC;
    int k = (kc << 5) + ((lane >> 4) << 4) + j;
    int n = (nTile << 4) + (lane & 15);
    P[idx] = (bf16_t)W[(size_t)k * N + n];
}

__global__ void f32_to_bf16_kernel(const float* __restrict__ src,
                                   bf16_t* __restrict__ dst, size_t n) {
    size_t i = (size_t)blockIdx.x * blockDim.x + threadIdx.x;
    if (i < n) dst[i] = (bf16_t)src[i];
}

__global__ void init_h_kernel(const float* __restrict__ carry,
                              float* __restrict__ hf,
                              bf16_t* __restrict__ hbf, int n) {
    int i = blockIdx.x * blockDim.x + threadIdx.x;
    if (i < n) { float v = carry[i]; hf[i] = v; hbf[i] = (bf16_t)v; }
}

// A-fragment: 16-bit A 16x32 layout (contiguous k, matches packed B):
//   lanes 0-15: M=lane,   K = k0+{0..15}
//   lanes16-31: M=lane-16,K = k0+{16..31}
// -> per lane: two contiguous 16B loads from the row-major bf16 row.
__device__ __forceinline__ v16bf load_a_frag(const bf16_t* __restrict__ arow,
                                             int k0, int kb_off) {
    union { v16bf v; v8bf h[2]; } a;
    a.h[0] = *(const v8bf*)(arow + k0 + kb_off);
    a.h[1] = *(const v8bf*)(arow + k0 + kb_off + 8);
    return a.v;
}

__device__ __forceinline__ float fast_sigmoid(float x) {
    return 1.0f / (1.0f + __expf(-x));
}
__device__ __forceinline__ float fast_tanh(float x) {
    float e = __expf(-2.0f * x);
    return (1.0f - e) / (1.0f + e);
}

// ---------------------------------------------------------------------------
// Phase 1: xr/xz/xn = x @ W_i{r,z,n} + b.  M = T*B (tiles of 16 share one t),
// each wave owns one 16x16 tile of one (mt, nt) and all 3 gates.
// proj layout: [T][B][H] f32 so each timestep reads a contiguous 256KB slab.
// ---------------------------------------------------------------------------
__global__ __launch_bounds__(256) void proj_kernel(
    const bf16_t* __restrict__ xbf,    // [B][T][D] bf16
    const bf16_t* __restrict__ pWi,    // 3 packed gates, stride D_*H_
    const float* __restrict__ b_ir, const float* __restrict__ b_iz,
    const float* __restrict__ b_in,
    float* __restrict__ projR, float* __restrict__ projZ,
    float* __restrict__ projN)
{
    const int lane = threadIdx.x & 31;
    const int wv   = threadIdx.x >> 5;
    const int w    = blockIdx.x * 8 + wv;
    const int NT   = H_ / 16;                 // 64 col tiles
    const int mt   = w / NT;
    const int nt   = w - mt * NT;
    const int m0   = mt * 16;                 // row index m = t*B + b
    const int t    = m0 >> 6;
    const int b0   = m0 & (B_ - 1);
    const int kb_off = (lane >> 4) << 4;

    const bf16_t* __restrict__ arow =
        xbf + ((size_t)(b0 + (lane & 15)) * T_ + t) * D_;

    const int KC = D_ / 32;
    const bf16_t* __restrict__ pR = pWi + ((size_t)nt * KC) * 512 + lane * 16;
    const bf16_t* __restrict__ pZ = pR + (size_t)D_ * H_;
    const bf16_t* __restrict__ pN = pZ + (size_t)D_ * H_;

    v8f accR = {}; v8f accZ = {}; v8f accN = {};

    // software-pipelined k-loop: prefetch kc+1 before the WMMAs of kc
    v16bf aC  = load_a_frag(arow, 0, kb_off);
    v16bf brC = *(const v16bf*)(pR);
    v16bf bzC = *(const v16bf*)(pZ);
    v16bf bnC = *(const v16bf*)(pN);
    for (int kc = 0; kc < KC - 1; ++kc) {
        v16bf aN  = load_a_frag(arow, (kc + 1) * 32, kb_off);
        v16bf brN = *(const v16bf*)(pR + (size_t)(kc + 1) * 512);
        v16bf bzN = *(const v16bf*)(pZ + (size_t)(kc + 1) * 512);
        v16bf bnN = *(const v16bf*)(pN + (size_t)(kc + 1) * 512);
        accR = __builtin_amdgcn_wmma_f32_16x16x32_bf16(false, aC, false, brC,
                                                       (short)0, accR, false, false);
        accZ = __builtin_amdgcn_wmma_f32_16x16x32_bf16(false, aC, false, bzC,
                                                       (short)0, accZ, false, false);
        accN = __builtin_amdgcn_wmma_f32_16x16x32_bf16(false, aC, false, bnC,
                                                       (short)0, accN, false, false);
        aC = aN; brC = brN; bzC = bzN; bnC = bnN;
    }
    accR = __builtin_amdgcn_wmma_f32_16x16x32_bf16(false, aC, false, brC,
                                                   (short)0, accR, false, false);
    accZ = __builtin_amdgcn_wmma_f32_16x16x32_bf16(false, aC, false, bzC,
                                                   (short)0, accZ, false, false);
    accN = __builtin_amdgcn_wmma_f32_16x16x32_bf16(false, aC, false, bnC,
                                                   (short)0, accN, false, false);

    // C/D layout: VGPR i -> row mbase+i, col n (lanes 0-15: M=0..7, 16-31: M=8..15)
    const int n     = nt * 16 + (lane & 15);
    const int mbase = m0 + ((lane >> 4) << 3);
    const float vr = b_ir[n], vz = b_iz[n], vn = b_in[n];
#pragma unroll
    for (int i = 0; i < 8; ++i) {
        size_t o = (size_t)(mbase + i) * H_ + n;   // (t*B + b)*H + n
        projR[o] = accR[i] + vr;
        projZ[o] = accZ[i] + vz;
        projN[o] = accN[i] + vn;
    }
}

// ---------------------------------------------------------------------------
// Phase 2: one launch per timestep. 256 tiles (4 Mtiles x 64 Ntiles),
// 8 waves/block, 32 blocks. Weights (6MB bf16) stay L2-resident.
// ---------------------------------------------------------------------------
__global__ __launch_bounds__(256) void gru_step_kernel(
    const bf16_t* __restrict__ hbf_in, const float* __restrict__ hf_in,
    bf16_t* __restrict__ hbf_out, float* __restrict__ hf_out,
    const bf16_t* __restrict__ pWh,    // 3 packed gates, stride H_*H_
    const float* __restrict__ b_hn,
    const float* __restrict__ projR, const float* __restrict__ projZ,
    const float* __restrict__ projN,
    float* __restrict__ out, int t)
{
    const int lane = threadIdx.x & 31;
    const int wv   = threadIdx.x >> 5;
    const int w    = blockIdx.x * 8 + wv;     // 0..255
    const int nt   = w & 63;
    const int mt   = w >> 6;
    const int m0   = mt * 16;                 // batch rows
    const int n0   = nt * 16;
    const int kb_off = (lane >> 4) << 4;

    const bf16_t* __restrict__ arow = hbf_in + (size_t)(m0 + (lane & 15)) * H_;

    const int KC = H_ / 32;
    const bf16_t* __restrict__ pR = pWh + ((size_t)nt * KC) * 512 + lane * 16;
    const bf16_t* __restrict__ pZ = pR + (size_t)H_ * H_;
    const bf16_t* __restrict__ pN = pZ + (size_t)H_ * H_;

    v8f accR = {}; v8f accZ = {}; v8f accN = {};

    v16bf aC  = load_a_frag(arow, 0, kb_off);
    v16bf brC = *(const v16bf*)(pR);
    v16bf bzC = *(const v16bf*)(pZ);
    v16bf bnC = *(const v16bf*)(pN);
    for (int kc = 0; kc < KC - 1; ++kc) {
        v16bf aN  = load_a_frag(arow, (kc + 1) * 32, kb_off);
        v16bf brN = *(const v16bf*)(pR + (size_t)(kc + 1) * 512);
        v16bf bzN = *(const v16bf*)(pZ + (size_t)(kc + 1) * 512);
        v16bf bnN = *(const v16bf*)(pN + (size_t)(kc + 1) * 512);
        accR = __builtin_amdgcn_wmma_f32_16x16x32_bf16(false, aC, false, brC,
                                                       (short)0, accR, false, false);
        accZ = __builtin_amdgcn_wmma_f32_16x16x32_bf16(false, aC, false, bzC,
                                                       (short)0, accZ, false, false);
        accN = __builtin_amdgcn_wmma_f32_16x16x32_bf16(false, aC, false, bnC,
                                                       (short)0, accN, false, false);
        aC = aN; brC = brN; bzC = bzN; bnC = bnN;
    }
    accR = __builtin_amdgcn_wmma_f32_16x16x32_bf16(false, aC, false, brC,
                                                   (short)0, accR, false, false);
    accZ = __builtin_amdgcn_wmma_f32_16x16x32_bf16(false, aC, false, bzC,
                                                   (short)0, accZ, false, false);
    accN = __builtin_amdgcn_wmma_f32_16x16x32_bf16(false, aC, false, bnC,
                                                   (short)0, accN, false, false);

    const int n     = n0 + (lane & 15);
    const int mbase = m0 + ((lane >> 4) << 3);
    const float bhn = b_hn[n];
    const size_t projbase = (size_t)t * B_ * H_;
#pragma unroll
    for (int i = 0; i < 8; ++i) {
        const int b   = mbase + i;
        const size_t hb = (size_t)b * H_ + n;
        const float xr = projR[projbase + hb];
        const float xz = projZ[projbase + hb];
        const float xn = projN[projbase + hb];
        const float r  = fast_sigmoid(xr + accR[i]);
        const float z  = fast_sigmoid(xz + accZ[i]);
        const float nv = fast_tanh(xn + r * (accN[i] + bhn));
        const float ho = hf_in[hb];
        const float hn = (1.0f - z) * nv + z * ho;
        hf_out[hb]  = hn;
        hbf_out[hb] = (bf16_t)hn;
        out[((size_t)b * T_ + t) * H_ + n] = hn;   // [B][T][H]
    }
}

// ---------------------------------------------------------------------------
extern "C" void kernel_launch(void* const* d_in, const int* in_sizes, int n_in,
                              void* d_out, int out_size, void* d_ws, size_t ws_size,
                              hipStream_t stream) {
    const float* carry = (const float*)d_in[0];
    const float* x     = (const float*)d_in[1];
    const float* W_ir  = (const float*)d_in[2];
    const float* W_iz  = (const float*)d_in[3];
    const float* W_in  = (const float*)d_in[4];
    const float* b_ir  = (const float*)d_in[5];
    const float* b_iz  = (const float*)d_in[6];
    const float* b_in  = (const float*)d_in[7];
    const float* W_hr  = (const float*)d_in[8];
    const float* W_hz  = (const float*)d_in[9];
    const float* W_hn  = (const float*)d_in[10];
    const float* b_hn  = (const float*)d_in[11];
    float* out = (float*)d_out;

    char* ws = (char*)d_ws;
    size_t off = 0;
    auto wsa = [&](size_t bytes) -> char* {
        char* p = ws + off;
        off += (bytes + 255) & ~(size_t)255;
        return p;
    };
    bf16_t* xbf   = (bf16_t*)wsa((size_t)B_ * T_ * D_ * sizeof(bf16_t));
    bf16_t* pWi   = (bf16_t*)wsa((size_t)3 * D_ * H_ * sizeof(bf16_t));
    bf16_t* pWh   = (bf16_t*)wsa((size_t)3 * H_ * H_ * sizeof(bf16_t));
    float*  projR = (float*) wsa((size_t)T_ * B_ * H_ * sizeof(float));
    float*  projZ = (float*) wsa((size_t)T_ * B_ * H_ * sizeof(float));
    float*  projN = (float*) wsa((size_t)T_ * B_ * H_ * sizeof(float));
    float*  hf0   = (float*) wsa((size_t)B_ * H_ * sizeof(float));
    float*  hf1   = (float*) wsa((size_t)B_ * H_ * sizeof(float));
    bf16_t* hb0   = (bf16_t*)wsa((size_t)B_ * H_ * sizeof(bf16_t));
    bf16_t* hb1   = (bf16_t*)wsa((size_t)B_ * H_ * sizeof(bf16_t));
    float*  hf[2]   = { hf0, hf1 };
    bf16_t* hbf[2]  = { hb0, hb1 };

    // --- pack all six weight matrices into WMMA fragment order (bf16) ---
    const int packBlocks = (D_ * H_ + 255) / 256;
    pack_w_kernel<<<packBlocks, 256, 0, stream>>>(W_ir, pWi + 0 * (size_t)D_ * H_, D_, H_);
    pack_w_kernel<<<packBlocks, 256, 0, stream>>>(W_iz, pWi + 1 * (size_t)D_ * H_, D_, H_);
    pack_w_kernel<<<packBlocks, 256, 0, stream>>>(W_in, pWi + 2 * (size_t)D_ * H_, D_, H_);
    pack_w_kernel<<<packBlocks, 256, 0, stream>>>(W_hr, pWh + 0 * (size_t)H_ * H_, H_, H_);
    pack_w_kernel<<<packBlocks, 256, 0, stream>>>(W_hz, pWh + 1 * (size_t)H_ * H_, H_, H_);
    pack_w_kernel<<<packBlocks, 256, 0, stream>>>(W_hn, pWh + 2 * (size_t)H_ * H_, H_, H_);

    // --- x -> bf16 ---
    const size_t nx = (size_t)B_ * T_ * D_;
    f32_to_bf16_kernel<<<(unsigned)((nx + 255) / 256), 256, 0, stream>>>(x, xbf, nx);

    // --- h0 ---
    init_h_kernel<<<(B_ * H_ + 255) / 256, 256, 0, stream>>>(carry, hf[0], hbf[0], B_ * H_);

    // --- input projections: (T*B/16) x (H/16) wave-tiles, 8 waves/block ---
    const int mtiles = (T_ * B_) / 16;       // 2048
    const int ntiles = H_ / 16;              // 64
    proj_kernel<<<(mtiles * ntiles) / 8, 256, 0, stream>>>(
        xbf, pWi, b_ir, b_iz, b_in, projR, projZ, projN);

    // --- recurrent scan: 512 sequential step kernels, h double-buffered ---
    for (int t = 0; t < T_; ++t) {
        const int src = t & 1, dst = src ^ 1;
        gru_step_kernel<<<32, 256, 0, stream>>>(
            hbf[src], hf[src], hbf[dst], hf[dst],
            pWh, b_hn, projR, projZ, projN, out, t);
    }
}
